// SearchCNN_28080496181708
// MI455X (gfx1250) — compile-verified, matrix operand order checked by reading, not run on hardware
//
#include <hip/hip_runtime.h>

#define NB 32
#define NA 8732
#define NG 16
#define NC 91
#define BA (NB * NA)

// CE pipeline geometry: 128-thread blocks (4 waves), 16 anchors per wave,
// 64 anchors per group -> one contiguous 23,296 B logits slab per group.
#define GRP_ANCH   64
#define GRP_ELEMS  (GRP_ANCH * NC)          // 5824 floats
#define NGRP       (BA / GRP_ANCH)          // 4366 (exact)
#define CE_BLOCKS  512

typedef __attribute__((ext_vector_type(16))) _Float16 v16h;
typedef __attribute__((ext_vector_type(8)))  float    v8f;
typedef __attribute__((ext_vector_type(4)))  unsigned v4u;
typedef __attribute__((ext_vector_type(8)))  int      v8i;
typedef __attribute__((ext_vector_type(4)))  int      v4i;

// ---------------------------------------------------------------- LDS helpers
__device__ __forceinline__ unsigned lds_offset_of(const void* p) {
  // generic -> LDS(addrspace 3) cast; AS3 pointers are 32-bit workgroup offsets
  return (unsigned)(unsigned long long)(__attribute__((address_space(3))) const char*)p;
}

// ---------------------------------------------------------------- TDM issue
// 1-row 2D descriptor: copy GRP_ELEMS contiguous f32 from global to LDS.
// clang-23 toolchain: 6-arg builtin (v4u, v8i, v4i, v4i, v8i, i32 cpol).
__device__ __forceinline__ void tdm_load_group(const float* gsrc, unsigned lds_off) {
  const unsigned long long ga = (unsigned long long)(uintptr_t)gsrc;
  v4u g0;
  g0[0] = 1u;                                                  // count=1, user mode
  g0[1] = lds_off;                                             // lds_addr (bytes)
  g0[2] = (unsigned)(ga & 0xFFFFFFFFu);                        // global_addr[31:0]
  g0[3] = (unsigned)((ga >> 32) & 0x01FFFFFFu) | 0x80000000u;  // addr[56:32] | type=2
  v8i g1;
  g1[0] = 0x00020000;                      // workgroup_mask=0, data_size=2 (4 B)
  g1[1] = (int)((unsigned)GRP_ELEMS << 16);// tensor_dim0[15:0] @ [31:16]
  g1[2] = 0x00010000;                      // tensor_dim0 hi=0, tensor_dim1=1
  g1[3] = (int)((unsigned)GRP_ELEMS << 16);// tensor_dim1 hi=0, tile_dim0=5824
  g1[4] = 0;                               // tile_dim1=0 (unused), tile_dim2=0
  g1[5] = (int)GRP_ELEMS;                  // tensor_dim0_stride lo
  g1[6] = 0;
  g1[7] = 0;                               // tensor_dim1_stride=0 (unused)
  const v4i z4 = {0, 0, 0, 0};             // groups 2/3: zeroed (<=2D tensor)
  const v8i z8 = {0, 0, 0, 0, 0, 0, 0, 0};
  __builtin_amdgcn_tensor_load_to_lds(g0, g1, z4, z4, z8, 0);
}

// ---------------------------------------------------------------- init
__global__ __launch_bounds__(128) void init_kernel(float* bbox_sum, float* ce_fg_sum,
                                                   int* fg_cnt, float* topk_sum) {
  const int t = threadIdx.x;
  if (t == 0) { *bbox_sum = 0.f; *ce_fg_sum = 0.f; }
  if (t < NB) { fg_cnt[t] = 0; topk_sum[t] = 0.f; }
}

// ---------------------------------------------------------------- SSD matching
__global__ __launch_bounds__(256) void match_kernel(const float* __restrict__ anchors,
                                                    const float* __restrict__ gt_boxes,
                                                    int* __restrict__ matched) {
  const int b = blockIdx.x;
  const int tid = threadIdx.x;
  __shared__ float g4[NG][4];
  __shared__ float garea[NG];
  __shared__ float sval[NG][257];
  __shared__ int   sidx[NG][257];
  __shared__ int   bestA[NG];

  if (tid < NG * 4) g4[tid >> 2][tid & 3] = gt_boxes[(long)b * NG * 4 + tid];
  __syncthreads();
  if (tid < NG) garea[tid] = (g4[tid][2] - g4[tid][0]) * (g4[tid][3] - g4[tid][1]);
  __syncthreads();

  float bv[NG]; int ba[NG];
#pragma unroll
  for (int g = 0; g < NG; ++g) { bv[g] = -1.0f; ba[g] = NA; }

  const float* anc = anchors + (long)b * NA * 4;
  int* mrow = matched + (long)b * NA;

  for (int a = tid; a < NA; a += 256) {
    const float x0 = anc[a * 4 + 0], y0 = anc[a * 4 + 1];
    const float x1 = anc[a * 4 + 2], y1 = anc[a * 4 + 3];
    const float aarea = (x1 - x0) * (y1 - y0);
    float best = -1.0f; int barg = 0;
#pragma unroll
    for (int g = 0; g < NG; ++g) {
      float iw = fminf(g4[g][2], x1) - fmaxf(g4[g][0], x0);
      float ih = fminf(g4[g][3], y1) - fmaxf(g4[g][1], y0);
      iw = fmaxf(iw, 0.f); ih = fmaxf(ih, 0.f);
      const float inter = iw * ih;
      const float iou = inter / (garea[g] + aarea - inter);
      if (iou > best) { best = iou; barg = g; }       // first max (ascending g)
      if (iou > bv[g]) { bv[g] = iou; ba[g] = a; }    // first max (ascending a)
    }
    mrow[a] = (best < 0.5f) ? -1 : barg;
  }
#pragma unroll
  for (int g = 0; g < NG; ++g) { sval[g][tid] = bv[g]; sidx[g][tid] = ba[g]; }
  __syncthreads();
  if (tid < NG) {
    float v = -1.f; int ai = NA;
    for (int t = 0; t < 256; ++t) {
      const float tv = sval[tid][t]; const int ti = sidx[tid][t];
      if (tv > v || (tv == v && ti < ai)) { v = tv; ai = ti; }
    }
    bestA[tid] = ai;
  }
  __syncthreads();
  if (tid == 0) {
    for (int g = 0; g < NG; ++g) mrow[bestA[g]] = g;  // force-match, last gt wins
  }
}

// ---------------------------------------------------------------- box regression loss
__global__ __launch_bounds__(256) void bbox_kernel(const float* __restrict__ breg,
                                                   const float* __restrict__ anchors,
                                                   const float* __restrict__ gt_boxes,
                                                   const int* __restrict__ matched,
                                                   float* __restrict__ bbox_sum,
                                                   int* __restrict__ fg_cnt) {
  const long i = (long)blockIdx.x * 256 + threadIdx.x;
  if (i >= BA) return;
  const int mm = matched[i];
  if (mm < 0) return;
  const int b = (int)(i / NA);
  const float* anc = anchors + i * 4;
  const float* g = gt_boxes + ((long)b * NG + mm) * 4;
  const float aw = anc[2] - anc[0], ah = anc[3] - anc[1];
  const float acx = anc[0] + 0.5f * aw, acy = anc[1] + 0.5f * ah;
  const float gw = g[2] - g[0], gh = g[3] - g[1];
  const float gcx = g[0] + 0.5f * gw, gcy = g[1] + 0.5f * gh;
  const float t[4] = { 10.f * (gcx - acx) / aw, 10.f * (gcy - acy) / ah,
                       5.f * __logf(gw / aw),   5.f * __logf(gh / ah) };
  const float* r = breg + i * 4;
  float s = 0.f;
#pragma unroll
  for (int j = 0; j < 4; ++j) {
    const float d = r[j] - t[j];
    const float ad = fabsf(d);
    s += (ad < 1.f) ? 0.5f * d * d : ad - 0.5f;
  }
  atomicAdd(bbox_sum, s);
  atomicAdd(&fg_cnt[b], 1);
}

// ---------------------------------------------------------------- CE: TDM + WMMA
// Persistent blocks double-buffer 23 KB logits slabs via the Tensor Data Mover
// (TENSORcnt), then compute log-sum-exp with v_wmma_f32_16x16x32_f16 row sums.
// A-fragment packing per ISA 16-bit 16x32 layout: lane l=m+16h holds anchor m,
// K = 32c + 16*(p>>3) + 8h + (p&7).
__global__ __launch_bounds__(128) void ce_kernel(const float* __restrict__ logits,
                                                 const int* __restrict__ matched,
                                                 const int* __restrict__ gt_labels,
                                                 unsigned* __restrict__ keys,
                                                 float* __restrict__ ce_fg_sum) {
  __shared__ float sbuf[2][GRP_ELEMS];     // 2 x 23,296 B double buffer
  __shared__ float s_sum[4][16];

  const int tid  = threadIdx.x;
  const int lane = tid & 31;
  const int wave = tid >> 5;
  const int m = lane & 15;
  const int h = lane >> 4;

  v16h ones;
#pragma unroll
  for (int p = 0; p < 16; ++p) ones[p] = (_Float16)1.0f;

  int cur = 0;
  const long g0i = blockIdx.x;
  if (tid == 0 && g0i < NGRP)
    tdm_load_group(logits + g0i * GRP_ELEMS, lds_offset_of(&sbuf[0][0]));

  for (long grp = g0i; grp < NGRP; grp += gridDim.x) {
    const long gnext = grp + gridDim.x;
    if (tid == 0) {
      if (gnext < NGRP) {
        tdm_load_group(logits + gnext * GRP_ELEMS, lds_offset_of(&sbuf[cur ^ 1][0]));
        __builtin_amdgcn_s_wait_tensorcnt(1);   // front buffer complete
      } else {
        __builtin_amdgcn_s_wait_tensorcnt(0);
      }
    }
    __syncthreads();                            // slab visible to all waves

    const float* rowL = &sbuf[cur][(wave * 16 + m) * NC];

    float x[3][16];
    float mx = -__builtin_inff();
#pragma unroll
    for (int c = 0; c < 3; ++c) {
#pragma unroll
      for (int p = 0; p < 16; ++p) {
        const int k = c * 32 + (p >> 3) * 16 + h * 8 + (p & 7);
        const float v = (k < NC) ? rowL[k] : -__builtin_inff();
        x[c][p] = v;
        mx = fmaxf(mx, v);
      }
    }
    mx = fmaxf(mx, __shfl_xor(mx, 16, 32));     // combine half-row lanes

    v8f acc = {};
#pragma unroll
    for (int c = 0; c < 3; ++c) {
      v16h afr;
#pragma unroll
      for (int p = 0; p < 16; ++p) afr[p] = (_Float16)__expf(x[c][p] - mx);
      acc = __builtin_amdgcn_wmma_f32_16x16x32_f16(false, afr, false, ones,
                                                   (short)0, acc, false, false);
    }

    // D[M][0]: lane 0 holds M=0..7, lane 16 holds M=8..15 (C/D layout)
    if (lane == 0) {
#pragma unroll
      for (int r = 0; r < 8; ++r) s_sum[wave][r] = acc[r];
    } else if (lane == 16) {
#pragma unroll
      for (int r = 0; r < 8; ++r) s_sum[wave][8 + r] = acc[r];
    }
    __syncthreads();

    if (h == 0) {
      const long idx = grp * GRP_ANCH + wave * 16 + m;
      const int b = (int)(idx / NA);
      const float sumexp = s_sum[wave][m];
      const int mm = matched[idx];
      const bool fg = mm >= 0;
      const int lbl = fg ? gt_labels[b * NG + mm] : 0;
      const float ce = __logf(sumexp) + mx - rowL[lbl];
      keys[idx] = fg ? 0u : __float_as_uint(fmaxf(ce, 0.0f));
      if (fg) atomicAdd(ce_fg_sum, ce);
    }
    __syncthreads();                            // reads done before buffer reuse
    cur ^= 1;
  }
}

// ---------------------------------------------------------------- hard-negative top-k sum
__global__ __launch_bounds__(256) void topk_kernel(const unsigned* __restrict__ keys,
                                                   const int* __restrict__ fg_cnt,
                                                   float* __restrict__ topk_sum) {
  const int b = blockIdx.x;
  const int tid = threadIdx.x;
  const unsigned* kk = keys + (long)b * NA;
  int k = 3 * fg_cnt[b];
  if (k > NA) k = NA;
  if (k == 0) { if (tid == 0) topk_sum[b] = 0.f; return; }

  __shared__ unsigned hist[256];
  __shared__ unsigned s_prefix;
  __shared__ int s_rem;
  __shared__ float red[256];

  unsigned prefix = 0;
  int rem = k;
  for (int shift = 24; shift >= 0; shift -= 8) {
    hist[tid] = 0;
    __syncthreads();
    const unsigned hi_mask = (shift == 24) ? 0u : (0xFFFFFFFFu << (shift + 8));
    for (int a = tid; a < NA; a += 256) {
      const unsigned key = kk[a];
      if ((key & hi_mask) == prefix)
        atomicAdd(&hist[(key >> shift) & 0xFFu], 1u);
    }
    __syncthreads();
    if (tid == 0) {
      int r = rem; unsigned byte = 0;
      for (int bin = 255; bin >= 0; --bin) {
        const int c = (int)hist[bin];
        if (r <= c) { byte = (unsigned)bin; break; }
        r -= c;
      }
      s_prefix = prefix | (byte << shift);
      s_rem = r;
    }
    __syncthreads();
    prefix = s_prefix;
    rem = s_rem;
    __syncthreads();
  }

  float partial = 0.f;
  for (int a = tid; a < NA; a += 256) {
    const unsigned key = kk[a];
    if (key > prefix) partial += __uint_as_float(key);
  }
  red[tid] = partial;
  __syncthreads();
  for (int s = 128; s > 0; s >>= 1) {
    if (tid < s) red[tid] += red[tid + s];
    __syncthreads();
  }
  if (tid == 0) topk_sum[b] = red[0] + (float)rem * __uint_as_float(prefix);
}

// ---------------------------------------------------------------- finalize
__global__ void finalize_kernel(const float* bbox_sum, const float* ce_fg_sum,
                                const float* topk_sum, const int* fg_cnt,
                                float* out) {
  if (threadIdx.x == 0) {
    int n = 0; float cls = *ce_fg_sum;
    for (int b = 0; b < NB; ++b) { n += fg_cnt[b]; cls += topk_sum[b]; }
    const float nf = fmaxf(1.0f, (float)n);
    out[0] = *bbox_sum / nf;
    out[1] = cls / nf;
  }
}

// ---------------------------------------------------------------- launcher
extern "C" void kernel_launch(void* const* d_in, const int* in_sizes, int n_in,
                              void* d_out, int out_size, void* d_ws, size_t ws_size,
                              hipStream_t stream) {
  (void)in_sizes; (void)n_in; (void)out_size; (void)ws_size;
  const float* cls_logits = (const float*)d_in[0];
  const float* bbox_reg   = (const float*)d_in[1];
  const float* anchors    = (const float*)d_in[2];
  const float* gt_boxes   = (const float*)d_in[3];
  const int*   gt_labels  = (const int*)d_in[4];

  char* ws = (char*)d_ws;
  int*      matched   = (int*)ws;      ws += (size_t)BA * sizeof(int);
  unsigned* keys      = (unsigned*)ws; ws += (size_t)BA * sizeof(unsigned);
  float*    bbox_sum  = (float*)ws;    ws += sizeof(float);
  float*    ce_fg_sum = (float*)ws;    ws += sizeof(float);
  float*    topk_sum  = (float*)ws;    ws += NB * sizeof(float);
  int*      fg_cnt    = (int*)ws;      ws += NB * sizeof(int);

  hipLaunchKernelGGL(init_kernel, dim3(1), dim3(128), 0, stream,
                     bbox_sum, ce_fg_sum, fg_cnt, topk_sum);
  hipLaunchKernelGGL(match_kernel, dim3(NB), dim3(256), 0, stream,
                     anchors, gt_boxes, matched);
  hipLaunchKernelGGL(bbox_kernel, dim3((BA + 255) / 256), dim3(256), 0, stream,
                     bbox_reg, anchors, gt_boxes, matched, bbox_sum, fg_cnt);
  hipLaunchKernelGGL(ce_kernel, dim3(CE_BLOCKS), dim3(128), 0, stream,
                     cls_logits, matched, gt_labels, keys, ce_fg_sum);
  hipLaunchKernelGGL(topk_kernel, dim3(NB), dim3(256), 0, stream,
                     keys, fg_cnt, topk_sum);
  hipLaunchKernelGGL(finalize_kernel, dim3(1), dim3(32), 0, stream,
                     bbox_sum, ce_fg_sum, topk_sum, fg_cnt, (float*)d_out);
}
